// DeepSeekRelationalModel_72808285601944
// MI455X (gfx1250) — compile-verified, matrix operand order checked by reading, not run on hardware
//
#include <hip/hip_runtime.h>
#include <hip/hip_bf16.h>

// ---------------- problem constants ----------------
#define T_TOK 4096   // B*S tokens
#define DDIM  2048   // hidden size
#define HDIM  1408   // moe intermediate
#define NEXP  8      // experts
// top_k is fixed to 2 by the reference

// GEMM tiling: 64x64 output tile, K-step 64, 128 threads (4 waves).
// Each wave owns a 16-row strip; per K-step it runs 2 A-fragments x 4 N-subtiles
// of v_wmma_f32_16x16x32_bf16. LDS tiles are double-buffered and filled with
// gfx1250 async global->LDS DMA so copies overlap WMMA math.
#define BM 64
#define BN 64
#define BK 64

typedef __attribute__((ext_vector_type(16))) __bf16 v16bf;
typedef __attribute__((ext_vector_type(8)))  __bf16 v8bf;
typedef __attribute__((ext_vector_type(8)))  float  v8f;
typedef __attribute__((ext_vector_type(4)))  int    v4i;

// ---------------- gfx1250 async global->LDS path ----------------
#if defined(__gfx1250__) && \
    __has_builtin(__builtin_amdgcn_global_load_async_to_lds_b128) && \
    __has_builtin(__builtin_amdgcn_s_wait_asynccnt)
#define ASYNC_COPY 1
#else
#define ASYNC_COPY 0
#endif

__device__ __forceinline__ void cp16(void* lds_dst, const void* gsrc) {
#if ASYNC_COPY
  // (AS1 v4i* src, AS3 v4i* dst, imm offset, imm cpol) — confirmed by round-1 diag
  __builtin_amdgcn_global_load_async_to_lds_b128(
      (__attribute__((address_space(1))) v4i*)gsrc,
      (__attribute__((address_space(3))) v4i*)lds_dst,
      0, 0);
#else
  *(uint4*)lds_dst = *(const uint4*)gsrc;
#endif
}

// Wait until at most N async copies remain in flight (per-wave ASYNCcnt).
#if ASYNC_COPY
#define CP_WAIT(n) __builtin_amdgcn_s_wait_asynccnt(n)
#else
#define CP_WAIT(n) ((void)0)
#endif

// ---------------- WMMA fragment loader ----------------
// Loads a 16x32 bf16 fragment from an LDS region laid out as [16 rows][BK cols]
// (pass base already offset to the 32-wide K-substep). CDNA5 16-bit A lane map
// (B staged as B^T rows, same map):
//   lane<16 : row = lane,    K in {0..7, 16..23}
//   lane>=16: row = lane-16, K in {8..15, 24..31}
__device__ __forceinline__ v16bf load_frag(const __bf16* base, int lane) {
  int row = lane & 15;
  int ko  = (lane < 16) ? 0 : 8;
  const v8bf lo = *(const v8bf*)(base + row * BK + ko);
  const v8bf hi = *(const v8bf*)(base + row * BK + ko + 16);
  v16bf r;
#pragma unroll
  for (int i = 0; i < 8; i++) { r[i] = lo[i]; r[i + 8] = hi[i]; }
  return r;
}

// ---------------- small utility kernels ----------------
__global__ void zero_counts_kernel(int* counts) {
  if (threadIdx.x < NEXP) counts[threadIdx.x] = 0;
}

__global__ void cast_bf16_kernel(const float* __restrict__ in,
                                 __bf16* __restrict__ out, size_t n) {
  size_t i = (size_t)blockIdx.x * blockDim.x + threadIdx.x;
  size_t stride = (size_t)gridDim.x * blockDim.x;
  for (; i < n; i += stride) out[i] = (__bf16)in[i];
}

__global__ void residual_kernel(const float* __restrict__ x,
                                float* __restrict__ out, size_t n) {
  size_t i = (size_t)blockIdx.x * blockDim.x + threadIdx.x;
  size_t stride = (size_t)gridDim.x * blockDim.x;
  for (; i < n; i += stride) out[i] = x[i];
}

__global__ void offs_kernel(const int* __restrict__ counts, int* __restrict__ offs) {
  if (threadIdx.x == 0) {
    int a = 0;
    for (int e = 0; e < NEXP; e++) { offs[e] = a; a += counts[e]; }
    offs[NEXP] = a;
  }
}

// ---------------- gating: softmax + top-2 routing ----------------
__global__ void gate_kernel(const float* __restrict__ x, const float* __restrict__ Wg,
                            int* __restrict__ counts, int* __restrict__ lists,
                            float* __restrict__ wts) {
  __shared__ float red[256];
  const int t = blockIdx.x, tid = threadIdx.x;

  float acc[NEXP];
#pragma unroll
  for (int e = 0; e < NEXP; e++) acc[e] = 0.f;
  for (int d = tid; d < DDIM; d += 256) {
    float xv = x[(size_t)t * DDIM + d];
#pragma unroll
    for (int e = 0; e < NEXP; e++) acc[e] += xv * Wg[e * DDIM + d];
  }

  float sc[NEXP];
  for (int e = 0; e < NEXP; e++) {
    red[tid] = acc[e];
    __syncthreads();
    for (int s = 128; s > 0; s >>= 1) {
      if (tid < s) red[tid] += red[tid + s];
      __syncthreads();
    }
    if (tid == 0) sc[e] = red[0];
    __syncthreads();
  }

  if (tid == 0) {
    float m = sc[0];
    for (int e = 1; e < NEXP; e++) m = fmaxf(m, sc[e]);
    float p[NEXP], sum = 0.f;
    for (int e = 0; e < NEXP; e++) { p[e] = __expf(sc[e] - m); sum += p[e]; }
    float inv = 1.f / sum;
    // top-2 (lowest index wins ties, like lax.top_k)
    int i0 = 0;
    for (int e = 1; e < NEXP; e++) if (sc[e] > sc[i0]) i0 = e;
    int i1 = (i0 == 0) ? 1 : 0;
    for (int e = 0; e < NEXP; e++) if (e != i0 && sc[e] > sc[i1]) i1 = e;

    int s0 = atomicAdd(&counts[i0], 1);
    lists[i0 * T_TOK + s0] = t; wts[i0 * T_TOK + s0] = p[i0] * inv;
    int s1 = atomicAdd(&counts[i1], 1);
    lists[i1 * T_TOK + s1] = t; wts[i1 * T_TOK + s1] = p[i1] * inv;
  }
}

// ---------------- phase A: h = silu(x W1^T) * (x W3^T), stored bf16 ----------------
// 12 async b128 copies per thread per K-step (A, B1, B3 tiles).
__global__ void __launch_bounds__(128)
expert_up_kernel(const __bf16* __restrict__ xb, const __bf16* __restrict__ w1b,
                 const __bf16* __restrict__ w3b, __bf16* __restrict__ hb,
                 const int* __restrict__ counts, const int* __restrict__ offs,
                 const int* __restrict__ lists) {
  __shared__ __bf16 As[2][BM * BK];
  __shared__ __bf16 B1s[2][BN * BK];
  __shared__ __bf16 B3s[2][BN * BK];
  __shared__ int toks[BM];

  const int e = blockIdx.z;
  const int cnt = counts[e];
  const int m_base = blockIdx.x * BM;
  if (m_base >= cnt) return;
  const int rows = min(BM, cnt - m_base);
  const int n0 = blockIdx.y * BN;
  const int tid = threadIdx.x;

  if (tid < BM) {
    int r = (tid < rows) ? tid : 0;
    toks[tid] = lists[e * T_TOK + m_base + r];
  }
  __syncthreads();

  const int wave = tid >> 5, lane = tid & 31;

  // per-thread chunk coordinates: 512 16B-chunks per tile, 4 per thread
  int crow[4], ccin[4];
#pragma unroll
  for (int i = 0; i < 4; i++) {
    int c = tid + i * 128;
    crow[i] = c >> 3;           // 8 chunks per 64-elem row
    ccin[i] = c & 7;
  }

  auto issue = [&](int buf, int k0) {
#pragma unroll
    for (int i = 0; i < 4; i++) {
      int row = crow[i], cin = ccin[i];
      cp16(&As[buf][row * BK + cin * 8],
           xb + (size_t)toks[row] * DDIM + k0 + cin * 8);
      cp16(&B1s[buf][row * BK + cin * 8],
           w1b + ((size_t)e * HDIM + n0 + row) * DDIM + k0 + cin * 8);
      cp16(&B3s[buf][row * BK + cin * 8],
           w3b + ((size_t)e * HDIM + n0 + row) * DDIM + k0 + cin * 8);
    }
  };

  v8f c1[4], c3[4];
#pragma unroll
  for (int j = 0; j < 4; j++)
#pragma unroll
    for (int g = 0; g < 8; g++) { c1[j][g] = 0.f; c3[j][g] = 0.f; }

  const int steps = DDIM / BK;   // 32
  issue(0, 0);
  for (int it = 0; it < steps; ++it) {
    const int cur = it & 1;
    if (it + 1 < steps) {
      issue(1 - cur, (it + 1) * BK);   // stream next tile behind the math
      CP_WAIT(12);                     // only next's 12 copies may remain
    } else {
      CP_WAIT(0);
    }
    __syncthreads();

    const __bf16* Abase = &As[cur][(wave * 16) * BK];
    v16bf a0 = load_frag(Abase, lane);        // K-substep 0 (cols 0..31)
    v16bf a1 = load_frag(Abase + 32, lane);   // K-substep 1 (cols 32..63)
#pragma unroll
    for (int j = 0; j < 4; j++) {
      const __bf16* b1base = &B1s[cur][(j * 16) * BK];
      const __bf16* b3base = &B3s[cur][(j * 16) * BK];
      v16bf b;
      b = load_frag(b1base, lane);
      c1[j] = __builtin_amdgcn_wmma_f32_16x16x32_bf16(false, a0, false, b,
                                                      (short)0, c1[j], false, false);
      b = load_frag(b1base + 32, lane);
      c1[j] = __builtin_amdgcn_wmma_f32_16x16x32_bf16(false, a1, false, b,
                                                      (short)0, c1[j], false, false);
      b = load_frag(b3base, lane);
      c3[j] = __builtin_amdgcn_wmma_f32_16x16x32_bf16(false, a0, false, b,
                                                      (short)0, c3[j], false, false);
      b = load_frag(b3base + 32, lane);
      c3[j] = __builtin_amdgcn_wmma_f32_16x16x32_bf16(false, a1, false, b,
                                                      (short)0, c3[j], false, false);
    }
    __syncthreads();   // done reading buf[cur]; it may be overwritten next it
  }

  const int hbase = offs[e] + m_base;
#pragma unroll
  for (int j = 0; j < 4; j++) {
#pragma unroll
    for (int g = 0; g < 8; g++) {
      int row = wave * 16 + g + ((lane >> 4) << 3);   // C layout: M = g + 8*(lane>=16)
      if (row < rows) {
        float u = c1[j][g];
        float sig = __builtin_amdgcn_rcpf(1.f + __expf(-u));   // fast sigmoid
        float h = u * sig * c3[j][g];                          // silu(u) * v
        hb[(size_t)(hbase + row) * HDIM + n0 + j * 16 + (lane & 15)] = (__bf16)h;
      }
    }
  }
}

// ---------------- phase B: out += wt * (h @ W2^T) ----------------
// 8 async b128 copies per thread per K-step (A, B tiles).
__global__ void __launch_bounds__(128)
expert_down_kernel(const __bf16* __restrict__ hb, const __bf16* __restrict__ w2b,
                   const int* __restrict__ counts, const int* __restrict__ offs,
                   const int* __restrict__ lists, const float* __restrict__ wts,
                   float* __restrict__ out) {
  __shared__ __bf16 As[2][BM * BK];
  __shared__ __bf16 Bs[2][BN * BK];
  __shared__ int toks[BM];
  __shared__ float wrow[BM];

  const int e = blockIdx.z;
  const int cnt = counts[e];
  const int m_base = blockIdx.x * BM;
  if (m_base >= cnt) return;
  const int rows = min(BM, cnt - m_base);
  const int n0 = blockIdx.y * BN;   // output D tile
  const int tid = threadIdx.x;

  if (tid < BM) {
    int r = (tid < rows) ? tid : 0;
    toks[tid] = lists[e * T_TOK + m_base + r];
    wrow[tid] = wts[e * T_TOK + m_base + r];
  }
  __syncthreads();

  const int wave = tid >> 5, lane = tid & 31;
  const int hbase = offs[e] + m_base;

  int crow[4], ccin[4], crr[4];
#pragma unroll
  for (int i = 0; i < 4; i++) {
    int c = tid + i * 128;
    crow[i] = c >> 3;
    ccin[i] = c & 7;
    crr[i]  = (crow[i] < rows) ? crow[i] : 0;   // clamp h reads in-bounds
  }

  auto issue = [&](int buf, int k0) {
#pragma unroll
    for (int i = 0; i < 4; i++) {
      int row = crow[i], cin = ccin[i];
      cp16(&As[buf][row * BK + cin * 8],
           hb + (size_t)(hbase + crr[i]) * HDIM + k0 + cin * 8);
      cp16(&Bs[buf][row * BK + cin * 8],
           w2b + ((size_t)e * DDIM + n0 + row) * HDIM + k0 + cin * 8);
    }
  };

  v8f c[4];
#pragma unroll
  for (int j = 0; j < 4; j++)
#pragma unroll
    for (int g = 0; g < 8; g++) c[j][g] = 0.f;

  const int steps = HDIM / BK;   // 22
  issue(0, 0);
  for (int it = 0; it < steps; ++it) {
    const int cur = it & 1;
    if (it + 1 < steps) {
      issue(1 - cur, (it + 1) * BK);
      CP_WAIT(8);
    } else {
      CP_WAIT(0);
    }
    __syncthreads();

    const __bf16* Abase = &As[cur][(wave * 16) * BK];
    v16bf a0 = load_frag(Abase, lane);
    v16bf a1 = load_frag(Abase + 32, lane);
#pragma unroll
    for (int j = 0; j < 4; j++) {
      const __bf16* bbase = &Bs[cur][(j * 16) * BK];
      v16bf b;
      b = load_frag(bbase, lane);
      c[j] = __builtin_amdgcn_wmma_f32_16x16x32_bf16(false, a0, false, b,
                                                     (short)0, c[j], false, false);
      b = load_frag(bbase + 32, lane);
      c[j] = __builtin_amdgcn_wmma_f32_16x16x32_bf16(false, a1, false, b,
                                                     (short)0, c[j], false, false);
    }
    __syncthreads();
  }

#pragma unroll
  for (int j = 0; j < 4; j++) {
#pragma unroll
    for (int g = 0; g < 8; g++) {
      int row = wave * 16 + g + ((lane >> 4) << 3);
      if (row < rows) {
        atomicAdd(&out[(size_t)toks[row] * DDIM + n0 + j * 16 + (lane & 15)],
                  wrow[row] * c[j][g]);
      }
    }
  }
}

// ---------------- host launcher ----------------
extern "C" void kernel_launch(void* const* d_in, const int* in_sizes, int n_in,
                              void* d_out, int out_size, void* d_ws, size_t ws_size,
                              hipStream_t stream) {
  (void)in_sizes; (void)n_in; (void)out_size; (void)ws_size;
  const float* x  = (const float*)d_in[0];   // [T, D]
  const float* Wg = (const float*)d_in[1];   // [E, D]
  const float* W1 = (const float*)d_in[2];   // [E, H, D]
  const float* W3 = (const float*)d_in[3];   // [E, H, D]
  const float* W2 = (const float*)d_in[4];   // [E, D, H]
  float* out = (float*)d_out;                // [T, D]

  char* ws = (char*)d_ws;
  size_t off = 0;
  auto take = [&](size_t bytes) -> char* {
    char* p = ws + off;
    off += (bytes + 255) & ~(size_t)255;
    return p;
  };
  __bf16* xb   = (__bf16*)take((size_t)T_TOK * DDIM * 2);
  __bf16* w1b  = (__bf16*)take((size_t)NEXP * HDIM * DDIM * 2);
  __bf16* w3b  = (__bf16*)take((size_t)NEXP * HDIM * DDIM * 2);
  __bf16* w2b  = (__bf16*)take((size_t)NEXP * DDIM * HDIM * 2);
  __bf16* hb   = (__bf16*)take((size_t)2 * T_TOK * HDIM * 2);
  int*    counts = (int*)take(NEXP * 4);
  int*    offs   = (int*)take((NEXP + 1) * 4);
  int*    lists  = (int*)take((size_t)NEXP * T_TOK * 4);
  float*  wts    = (float*)take((size_t)NEXP * T_TOK * 4);

  zero_counts_kernel<<<1, 32, 0, stream>>>(counts);
  cast_bf16_kernel<<<2048, 256, 0, stream>>>(x,  xb,  (size_t)T_TOK * DDIM);
  cast_bf16_kernel<<<4096, 256, 0, stream>>>(W1, w1b, (size_t)NEXP * HDIM * DDIM);
  cast_bf16_kernel<<<4096, 256, 0, stream>>>(W3, w3b, (size_t)NEXP * HDIM * DDIM);
  cast_bf16_kernel<<<4096, 256, 0, stream>>>(W2, w2b, (size_t)NEXP * DDIM * HDIM);
  gate_kernel<<<T_TOK, 256, 0, stream>>>(x, Wg, counts, lists, wts);
  offs_kernel<<<1, 32, 0, stream>>>(counts, offs);
  residual_kernel<<<2048, 256, 0, stream>>>(x, out, (size_t)T_TOK * DDIM);

  dim3 gA(T_TOK / BM, HDIM / BN, NEXP);   // 64 x 22 x 8, early-exit beyond counts[e]
  expert_up_kernel<<<gA, 128, 0, stream>>>(xb, w1b, w3b, hb, counts, offs, lists);

  dim3 gB(T_TOK / BM, DDIM / BN, NEXP);   // 64 x 32 x 8
  expert_down_kernel<<<gB, 128, 0, stream>>>(hb, w2b, counts, offs, lists, wts, out);
}
